// MultiHeadAttention_88338887344199
// MI455X (gfx1250) — matrix operand order repaired and checked by reading, not compiled, not run on hardware
//
#include <hip/hip_runtime.h>
#include <hip/hip_bf16.h>

// ---------------- problem constants ----------------
constexpr int LSEQ   = 2048;   // sequence length
constexpr int DMODEL = 2048;   // model dim
constexpr int NH     = 16;     // heads
constexpr int DKH    = 128;    // head dim (dk == dv)
constexpr float LN_EPS = 1e-5f;

typedef __bf16 bf16_t;
typedef __attribute__((ext_vector_type(16))) __bf16 v16bf;
typedef __attribute__((ext_vector_type(8)))  float  v8f;

// ---------------- WMMA fragment helpers (wave32, 16x16x32 bf16) ----------------
// A operand: 16x32 (MxK). lane 0-15: M=lane, K base 0; lane 16-31: M=lane-16, K base 8.
// vector elem i<8 -> K = kb+i ; elem i>=8 -> K = kb+16+(i-8).
__device__ __forceinline__ v16bf load_a_f32(const float* __restrict__ A, int lda, int m0, int k0) {
  const int lane = threadIdx.x & 31;
  const int m  = m0 + (lane & 15);
  const int kb = k0 + ((lane >> 4) << 3);
  const float* p = A + (size_t)m * lda + kb;
  v16bf a;
#pragma unroll
  for (int i = 0; i < 8; ++i) a[i] = (__bf16)p[i];
#pragma unroll
  for (int i = 0; i < 8; ++i) a[8 + i] = (__bf16)p[16 + i];
  return a;
}

__device__ __forceinline__ v16bf load_a_bf16(const bf16_t* __restrict__ A, int lda, int m0, int k0) {
  const int lane = threadIdx.x & 31;
  const int m  = m0 + (lane & 15);
  const int kb = k0 + ((lane >> 4) << 3);
  const bf16_t* p = A + (size_t)m * lda + kb;
  v16bf a;
#pragma unroll
  for (int i = 0; i < 8; ++i) a[i] = p[i];
#pragma unroll
  for (int i = 0; i < 8; ++i) a[8 + i] = p[16 + i];
  return a;
}

// B operand: 32x16 (KxN), data supplied as Brow[n][k] (row-major in k).
// lane 0-15: N=lane, K base 0; lane 16-31: N=lane-16, K base 8. elem i<8 -> K=kb+i ; elem i>=8 -> K=kb+16+(i-8).
__device__ __forceinline__ v16bf load_b_rows(const bf16_t* __restrict__ Bt, int ldb, int n0, int k0) {
  const int lane = threadIdx.x & 31;
  const int n  = n0 + (lane & 15);
  const int kb = k0 + ((lane >> 4) << 3);
  const bf16_t* p = Bt + (size_t)n * ldb + kb;
  v16bf b;
#pragma unroll
  for (int i = 0; i < 8; ++i) b[i] = p[i];
#pragma unroll
  for (int i = 0; i < 8; ++i) b[8 + i] = p[16 + i];
  return b;
}

__device__ __forceinline__ v8f wmma_bf16(v16bf a, v16bf b, v8f c) {
  return __builtin_amdgcn_wmma_f32_16x16x32_bf16(false, a, false, b, (short)0, c, false, false);
}

// C/D: lane 0-15: N=lane, VGPR r -> M=r ; lane 16-31: N=lane-16, VGPR r -> M=8+r.
__device__ __forceinline__ void store_c_f32(float* __restrict__ C, int ldc, int m0, int n0,
                                            v8f acc, float scale) {
  const int lane = threadIdx.x & 31;
  const int n  = n0 + (lane & 15);
  const int mb = m0 + ((lane >> 4) << 3);
#pragma unroll
  for (int r = 0; r < 8; ++r) C[(size_t)(mb + r) * ldc + n] = acc[r] * scale;
}

__device__ __forceinline__ void store_c_bf16(bf16_t* __restrict__ C, int ldc, int m0, int n0, v8f acc) {
  const int lane = threadIdx.x & 31;
  const int n  = n0 + (lane & 15);
  const int mb = m0 + ((lane >> 4) << 3);
#pragma unroll
  for (int r = 0; r < 8; ++r) C[(size_t)(mb + r) * ldc + n] = (__bf16)acc[r];
}

// store transposed: element (m,n) -> C[n][m]
__device__ __forceinline__ void store_c_bf16_T(bf16_t* __restrict__ C, int ldc, int m0, int n0, v8f acc) {
  const int lane = threadIdx.x & 31;
  const int n  = n0 + (lane & 15);
  const int mb = m0 + ((lane >> 4) << 3);
#pragma unroll
  for (int r = 0; r < 8; ++r) C[(size_t)n * ldc + (mb + r)] = (__bf16)acc[r];
}

// ---------------- kernel 0: weight transpose + f32 -> bf16 ----------------
// WqT/WkT/WvT: [H][DKH][DMODEL] bf16 ; projT: [DMODEL][H*DV] bf16
__global__ void prep_weights(const float* __restrict__ Wq, const float* __restrict__ Wk,
                             const float* __restrict__ Wv, const float* __restrict__ pw,
                             bf16_t* __restrict__ WqT, bf16_t* __restrict__ WkT,
                             bf16_t* __restrict__ WvT, bf16_t* __restrict__ projT) {
  const unsigned tid = blockIdx.x * blockDim.x + threadIdx.x;   // 0 .. 4*2^22-1
  const unsigned seg = tid >> 22;
  const unsigned r   = tid & 0x3FFFFFu;
  if (seg < 3) {
    const unsigned h  = r >> 18;            // / (DMODEL*DKH)
    const unsigned rr = r & 0x3FFFFu;
    const unsigned kd = rr >> 7;            // model-dim index
    const unsigned n  = rr & 127u;          // head-dim index
    const float* src = (seg == 0) ? Wq : (seg == 1) ? Wk : Wv;
    bf16_t* dst = (seg == 0) ? WqT : (seg == 1) ? WkT : WvT;
    dst[((size_t)(h * DKH + n)) * DMODEL + kd] = (__bf16)src[r];
  } else {
    const unsigned kd = r >> 11;            // H*DV index
    const unsigned n  = r & 2047u;          // model-dim index
    projT[(size_t)n * (NH * DKH) + kd] = (__bf16)pw[r];
  }
}

// ---------------- kernel 1: QKV projections (2x2 register-blocked) ----------------
// qh, kh: [H][L][DKH] bf16 ; vhT: [H][DKH][L] bf16
__global__ void qkv_proj(const float* __restrict__ q, const float* __restrict__ k,
                         const float* __restrict__ v,
                         const bf16_t* __restrict__ WqT, const bf16_t* __restrict__ WkT,
                         const bf16_t* __restrict__ WvT,
                         bf16_t* __restrict__ qh, bf16_t* __restrict__ kh,
                         bf16_t* __restrict__ vhT) {
  const int wave = blockIdx.x * 8 + (threadIdx.x >> 5);  // 0 .. 3*NH*64*4 - 1
  const int s    = wave >> 12;                            // / (NH*256)
  const int rem  = wave & 4095;
  const int h    = rem >> 8;
  const int t    = rem & 255;
  const int m0   = (t >> 2) * 32;      // 64 row-tiles of 32
  const int n0   = (t & 3) * 32;       // 4 col-tiles of 32

  const float*  X  = (s == 0) ? q : (s == 1) ? k : v;
  const bf16_t* WT = ((s == 0) ? WqT : (s == 1) ? WkT : WvT) + (size_t)h * DKH * DMODEL;

  v8f c00 = {}, c01 = {}, c10 = {}, c11 = {};
#pragma unroll 2
  for (int kk = 0; kk < DMODEL; kk += 32) {
    v16bf a0 = load_a_f32(X, DMODEL, m0, kk);
    v16bf a1 = load_a_f32(X, DMODEL, m0 + 16, kk);
    v16bf b0 = load_b_rows(WT, DMODEL, n0, kk);
    v16bf b1 = load_b_rows(WT, DMODEL, n0 + 16, kk);
    c00 = wmma_bf16(a0, b0, c00);
    c01 = wmma_bf16(a0, b1, c01);
    c10 = wmma_bf16(a1, b0, c10);
    c11 = wmma_bf16(a1, b1, c11);
  }
  if (s < 2) {
    bf16_t* Y = ((s == 0) ? qh : kh) + (size_t)h * LSEQ * DKH;
    store_c_bf16(Y, DKH, m0,      n0,      c00);
    store_c_bf16(Y, DKH, m0,      n0 + 16, c01);
    store_c_bf16(Y, DKH, m0 + 16, n0,      c10);
    store_c_bf16(Y, DKH, m0 + 16, n0 + 16, c11);
  } else {
    bf16_t* Y = vhT + (size_t)h * DKH * LSEQ;
    store_c_bf16_T(Y, LSEQ, m0,      n0,      c00);
    store_c_bf16_T(Y, LSEQ, m0,      n0 + 16, c01);
    store_c_bf16_T(Y, LSEQ, m0 + 16, n0,      c10);
    store_c_bf16_T(Y, LSEQ, m0 + 16, n0 + 16, c11);
  }
}

// ---------------- kernel 2: scores = qh * kh^T / sqrt(DMODEL) (2x2 blocked) ----------------
__global__ void scores_kernel(const bf16_t* __restrict__ qh, const bf16_t* __restrict__ kh,
                              float* __restrict__ attn) {
  const int wave = blockIdx.x * 8 + (threadIdx.x >> 5);  // 0 .. NH*64*64 - 1
  const int h  = wave >> 12;
  const int t  = wave & 4095;
  const int m0 = (t >> 6) * 32;    // query tile base
  const int n0 = (t & 63) * 32;    // key tile base

  const bf16_t* Q = qh + (size_t)h * LSEQ * DKH;
  const bf16_t* K = kh + (size_t)h * LSEQ * DKH;   // B operand: lane=key row, K-dim contiguous

  v8f c00 = {}, c01 = {}, c10 = {}, c11 = {};
#pragma unroll
  for (int kk = 0; kk < DKH; kk += 32) {
    v16bf a0 = load_a_bf16(Q, DKH, m0, kk);
    v16bf a1 = load_a_bf16(Q, DKH, m0 + 16, kk);
    v16bf b0 = load_b_rows(K, DKH, n0, kk);
    v16bf b1 = load_b_rows(K, DKH, n0 + 16, kk);
    c00 = wmma_bf16(a0, b0, c00);
    c01 = wmma_bf16(a0, b1, c01);
    c10 = wmma_bf16(a1, b0, c10);
    c11 = wmma_bf16(a1, b1, c11);
  }
  const float scale = 0.022097086912079608f;       // 1/sqrt(2048)
  float* S = attn + (size_t)h * LSEQ * LSEQ;
  store_c_f32(S, LSEQ, m0,      n0,      c00, scale);
  store_c_f32(S, LSEQ, m0,      n0 + 16, c01, scale);
  store_c_f32(S, LSEQ, m0 + 16, n0,      c10, scale);
  store_c_f32(S, LSEQ, m0 + 16, n0 + 16, c11, scale);
}

// ---------------- kernel 3: softmax over the QUERY axis (dim=1), in place ----------------
// grid: NH * 32 blocks ; 256 threads = 64 columns x 4 row-chunks (512 rows each)
__global__ void col_softmax(float* __restrict__ attn) {
  __shared__ float sm[256];
  const int h   = blockIdx.x >> 5;
  const int cb  = blockIdx.x & 31;
  const int c   = threadIdx.x & 63;          // column within this block
  const int rc  = threadIdx.x >> 6;          // row chunk 0..3
  const int col = cb * 64 + c;
  float* S = attn + (size_t)h * LSEQ * LSEQ;
  const int r0 = rc * (LSEQ / 4), r1 = r0 + (LSEQ / 4);

  float mx = -3.4e38f;
  for (int qi = r0; qi < r1; ++qi) mx = fmaxf(mx, S[(size_t)qi * LSEQ + col]);
  sm[threadIdx.x] = mx;
  __syncthreads();
  if (rc == 0) sm[c] = fmaxf(fmaxf(sm[c], sm[c + 64]), fmaxf(sm[c + 128], sm[c + 192]));
  __syncthreads();
  mx = sm[c];
  __syncthreads();

  float sum = 0.f;
  for (int qi = r0; qi < r1; ++qi) sum += __expf(S[(size_t)qi * LSEQ + col] - mx);
  sm[threadIdx.x] = sum;
  __syncthreads();
  if (rc == 0) sm[c] = (sm[c] + sm[c + 64]) + (sm[c + 128] + sm[c + 192]);
  __syncthreads();
  const float inv = __frcp_rn(sm[c]);

  for (int qi = r0; qi < r1; ++qi) {
    const size_t idx = (size_t)qi * LSEQ + col;
    S[idx] = __expf(S[idx] - mx) * inv;
  }
}

// ---------------- kernel 4: O[:, h*DV + v] = attn_h @ vh_h (2x2 blocked) ----------------
__global__ void attn_v(const float* __restrict__ attn, const bf16_t* __restrict__ vhT,
                       float* __restrict__ O) {
  const int wave = blockIdx.x * 8 + (threadIdx.x >> 5);  // 0 .. NH*64*4 - 1
  const int h  = wave >> 8;
  const int t  = wave & 255;
  const int m0 = (t >> 2) * 32;   // query tile base
  const int n0 = (t & 3) * 32;    // dv tile base

  const float*  A = attn + (size_t)h * LSEQ * LSEQ;
  const bf16_t* B = vhT  + (size_t)h * DKH * LSEQ;

  v8f c00 = {}, c01 = {}, c10 = {}, c11 = {};
#pragma unroll 2
  for (int kk = 0; kk < LSEQ; kk += 32) {
    v16bf a0 = load_a_f32(A, LSEQ, m0, kk);
    v16bf a1 = load_a_f32(A, LSEQ, m0 + 16, kk);
    v16bf b0 = load_b_rows(B, LSEQ, n0, kk);
    v16bf b1 = load_b_rows(B, LSEQ, n0 + 16, kk);
    c00 = wmma_bf16(a0, b0, c00);
    c01 = wmma_bf16(a0, b1, c01);
    c10 = wmma_bf16(a1, b0, c10);
    c11 = wmma_bf16(a1, b1, c11);
  }
  const int nb = h * DKH + n0;
  store_c_f32(O, NH * DKH, m0,      nb,      c00, 1.0f);
  store_c_f32(O, NH * DKH, m0,      nb + 16, c01, 1.0f);
  store_c_f32(O, NH * DKH, m0 + 16, nb,      c10, 1.0f);
  store_c_f32(O, NH * DKH, m0 + 16, nb + 16, c11, 1.0f);
}

// ---------------- kernel 5: y_pre = O @ proj_w + proj_b + residual(q) (2x2 blocked) ----------------
__device__ __forceinline__ void store_proj(float* __restrict__ yout, const float* __restrict__ proj_b,
                                           const float* __restrict__ resid, int m0, int n0, v8f acc) {
  const int lane = threadIdx.x & 31;
  const int n  = n0 + (lane & 15);
  const int mb = m0 + ((lane >> 4) << 3);
  const float bias = proj_b[n];
#pragma unroll
  for (int r = 0; r < 8; ++r) {
    const size_t idx = (size_t)(mb + r) * DMODEL + n;
    yout[idx] = acc[r] + bias + resid[idx];
  }
}

__global__ void proj_residual(const float* __restrict__ O, const bf16_t* __restrict__ projT,
                              const float* __restrict__ proj_b, const float* __restrict__ resid,
                              float* __restrict__ yout) {
  const int wave = blockIdx.x * 8 + (threadIdx.x >> 5);  // 0 .. 64*64 - 1
  const int m0 = (wave >> 6) * 32;
  const int n0 = (wave & 63) * 32;

  v8f c00 = {}, c01 = {}, c10 = {}, c11 = {};
#pragma unroll 2
  for (int kk = 0; kk < NH * DKH; kk += 32) {
    v16bf a0 = load_a_f32(O, NH * DKH, m0, kk);
    v16bf a1 = load_a_f32(O, NH * DKH, m0 + 16, kk);
    v16bf b0 = load_b_rows(projT, NH * DKH, n0, kk);
    v16bf b1 = load_b_rows(projT, NH * DKH, n0 + 16, kk);
    c00 = wmma_bf16(a0, b0, c00);
    c01 = wmma_bf16(a0, b1, c01);
    c10 = wmma_bf16(a1, b0, c10);
    c11 = wmma_bf16(a1, b1, c11);
  }
  store_proj(yout, proj_b, resid, m0,      n0,      c00);
  store_proj(yout, proj_b, resid, m0,      n0 + 16, c01);
  store_proj(yout, proj_b, resid, m0 + 16, n0,      c10);
  store_proj(yout, proj_b, resid, m0 + 16, n0 + 16, c11);
}

// ---------------- kernel 6: in-place row layernorm ----------------
__global__ void layernorm_rows(float* __restrict__ y, const float* __restrict__ g,
                               const float* __restrict__ b) {
  __shared__ float s1[256];
  __shared__ float s2[256];
  const int tid = threadIdx.x;
  float* x = y + (size_t)blockIdx.x * DMODEL;
  float sum = 0.f, sq = 0.f;
  for (int i = tid; i < DMODEL; i += 256) { const float v = x[i]; sum += v; sq += v * v; }
  s1[tid] = sum; s2[tid] = sq;
  __syncthreads();
  for (int off = 128; off > 0; off >>= 1) {
    if (tid < off) { s1[tid] += s1[tid + off]; s2[tid] += s2[tid + off]; }
    __syncthreads();
  }
  const float mean = s1[0] * (1.0f / DMODEL);
  const float var  = s2[0] * (1.0f / DMODEL) - mean * mean;
  const float rstd = rsqrtf(var + LN_EPS);
  for (int i = tid; i < DMODEL; i += 256) {
    x[i] = (x[i] - mean) * rstd * g[i] + b[i];
  }
}

// ---------------- host launcher ----------------
extern "C" void kernel_launch(void* const* d_in, const int* in_sizes, int n_in,
                              void* d_out, int out_size, void* d_ws, size_t ws_size,
                              hipStream_t stream) {
  (void)in_sizes; (void)n_in; (void)out_size; (void)ws_size;
  const float* q      = (const float*)d_in[0];
  const float* k      = (const float*)d_in[1];
  const float* v      = (const float*)d_in[2];
  const float* Wq     = (const float*)d_in[3];
  const float* Wk     = (const float*)d_in[4];
  const float* Wv     = (const float*)d_in[5];
  const float* pw     = (const float*)d_in[6];
  const float* pb     = (const float*)d_in[7];
  const float* ln_g   = (const float*)d_in[8];
  const float* ln_b   = (const float*)d_in[9];

  float* yout = (float*)d_out;                                  // (L, DMODEL)
  float* attn = (float*)d_out + (size_t)LSEQ * DMODEL;          // (H, L, L)

  // workspace carve-up
  char* w = (char*)d_ws;
  const size_t SZ_HEAD_BF16 = (size_t)NH * LSEQ * DKH * sizeof(bf16_t);   // 8 MiB each
  const size_t SZ_PROJ_BF16 = (size_t)DMODEL * NH * DKH * sizeof(bf16_t); // 8 MiB
  bf16_t* qh    = (bf16_t*)w; w += SZ_HEAD_BF16;
  bf16_t* kh    = (bf16_t*)w; w += SZ_HEAD_BF16;
  bf16_t* vhT   = (bf16_t*)w; w += SZ_HEAD_BF16;
  bf16_t* WqT   = (bf16_t*)w; w += SZ_HEAD_BF16;
  bf16_t* WkT   = (bf16_t*)w; w += SZ_HEAD_BF16;
  bf16_t* WvT   = (bf16_t*)w; w += SZ_HEAD_BF16;
  bf16_t* projT = (bf16_t*)w; w += SZ_PROJ_BF16;
  float*  O     = (float*)w;  w += (size_t)LSEQ * DMODEL * sizeof(float); // 16 MiB

  // 0) weights -> bf16 transposed
  prep_weights<<<65536, 256, 0, stream>>>(Wq, Wk, Wv, pw, WqT, WkT, WvT, projT);
  // 1) head projections (3*16 GEMMs of 2048x2048x128), 32x32 per wave
  qkv_proj<<<(3 * NH * 64 * 4) / 8, 256, 0, stream>>>(q, k, v, WqT, WkT, WvT, qh, kh, vhT);
  // 2) scores -> d_out attn region (scaled by 1/sqrt(DMODEL)), 32x32 per wave
  scores_kernel<<<(NH * 64 * 64) / 8, 256, 0, stream>>>(qh, kh, attn);
  // 3) softmax over query axis, in place
  col_softmax<<<NH * 32, 256, 0, stream>>>(attn);
  // 4) O = attn @ vh (concat over heads), 32x32 per wave
  attn_v<<<(NH * 64 * 4) / 8, 256, 0, stream>>>(attn, vhT, O);
  // 5) projection + bias + residual -> d_out y region, 32x32 per wave
  proj_residual<<<(64 * 64) / 8, 256, 0, stream>>>(O, projT, pb, q, yout);
  // 6) layernorm in place
  layernorm_rows<<<LSEQ, 256, 0, stream>>>(yout, ln_g, ln_b);
}
